// Attention_InfLevel_75814762709298
// MI455X (gfx1250) — compile-verified, matrix operand order checked by reading, not run on hardware
//
#include <hip/hip_runtime.h>
#include <hip/hip_bf16.h>

typedef __attribute__((ext_vector_type(16))) __bf16   v16bf;
typedef __attribute__((ext_vector_type(8)))  float    v8f;
typedef __attribute__((ext_vector_type(8)))  unsigned v8u;

#define NB 32
#define NN 1024
#define NM 1024
#define ND 128
#define GT_STRIDE 520   // 512 + 8 bf16 pad -> rows land on different LDS banks
#define NEG_BIG -9.0e15f
#define MSTRIDE 17      // mask words per thread, padded (gcd(17,64)=1 -> conflict-free)

// round-to-nearest-even f32 -> bf16 (bits)
__device__ __forceinline__ unsigned f2bf1(float f) {
  unsigned u = __builtin_bit_cast(unsigned, f);
  u += 0x7FFFu + ((u >> 16) & 1u);
  return u >> 16;
}
__device__ __forceinline__ unsigned f2bf2(float lo, float hi) {
  return f2bf1(lo) | (f2bf1(hi) << 16);
}
__device__ __forceinline__ float lrelu02(float s) {
  return s > 0.0f ? s : 0.2f * s;
}

// ---------------------------------------------------------------------------
// Kernel 1: e1[b,n] = h[b,n,:]·a1 ; e2[b,m] = g[b,m,:]·a2
// One wave (32 lanes x float4 = 128 = D) per row, shuffle-xor reduction.
// ---------------------------------------------------------------------------
__global__ __launch_bounds__(256) void gat_edge_kernel(
    const float* __restrict__ h, const float* __restrict__ g,
    const float* __restrict__ a1, const float* __restrict__ a2,
    float* __restrict__ e1, float* __restrict__ e2)
{
  const int tid  = threadIdx.x;
  const int lane = tid & 31;
  const int wave = tid >> 5;
  const int r    = blockIdx.x * 8 + wave;      // 0..65535

  const float* src;
  const float* avec;
  float* dst;
  if (r < NB * NN) { src = h + (size_t)r * ND;              avec = a1; dst = e1 + r; }
  else             { src = g + (size_t)(r - NB * NN) * ND;  avec = a2; dst = e2 + (r - NB * NN); }

  float4 x = *(const float4*)(src  + lane * 4);
  float4 w = *(const float4*)(avec + lane * 4);
  float v = x.x * w.x + x.y * w.y + x.z * w.z + x.w * w.w;
  v += __shfl_xor(v, 16);
  v += __shfl_xor(v, 8);
  v += __shfl_xor(v, 4);
  v += __shfl_xor(v, 2);
  v += __shfl_xor(v, 1);
  if (lane == 0) *dst = v;
}

// ---------------------------------------------------------------------------
// Kernel 2: fused mask + softmax*deg + (P @ g) with bf16 WMMA.
// Grid: (N/128, B). 256 threads = 8 waves, each wave owns 16 rows (one M-tile).
// Softmax without max-subtraction (shift-invariant; scores are small), so adj
// is read exactly once and only two exp evaluations per element are needed.
// Mask bits live in LDS (per-thread slots) so all loops stay rolled.
// ---------------------------------------------------------------------------
__global__ __launch_bounds__(256) void gat_attn_kernel(
    const float* __restrict__ g, const int* __restrict__ adj,
    const float* __restrict__ e1g, const float* __restrict__ e2g,
    float* __restrict__ out)
{
  __shared__ unsigned short gt[ND * GT_STRIDE]; // g chunk, transposed [d][m], bf16 (~130KB)
  __shared__ float e2s[NM];                     // e2 row for this batch (4KB)
  __shared__ unsigned mlds[256 * MSTRIDE];      // per-thread mask words (17KB)

  const int b    = blockIdx.y;
  const int tile = blockIdx.x;
  const int tid  = threadIdx.x;
  const int lane = tid & 31;
  const int wave = tid >> 5;
  const int rit  = lane & 15;        // row within wave's 16-row tile (A: lane -> M)
  const int half = lane >> 4;        // 0: K 0..7/16..23   1: K 8..15/24..31
  const int moff = half << 3;
  const int n0   = tile * 128 + wave * 16;
  const int row  = n0 + rit;
  const int mbase = tid * MSTRIDE;

  for (int i = tid; i < NM; i += 256) e2s[i] = e2g[b * NM + i];
  __syncthreads();

  const float e1v = e1g[b * NN + row];
  const int* adjrow = adj + (((size_t)b * NN + row) * NM);

  // ---- Pass A: single read of adj -> mask bits, degree, sum of exp ---------
  float sum = 0.0f;
  int   cnt = 0;
#pragma unroll 1
  for (int w = 0; w < 16; ++w) {                // one 32-bit mask word / iter
    unsigned wordbits = 0u;
#pragma unroll
    for (int kh = 0; kh < 2; ++kh) {
      const int k  = w * 2 + kh;
      const int g0 = k * 32 + moff;
      const int g1 = g0 + 16;
      const int4 aa = *(const int4*)(adjrow + g0);
      const int4 ab = *(const int4*)(adjrow + g0 + 4);
      const int4 ac = *(const int4*)(adjrow + g1);
      const int4 ad = *(const int4*)(adjrow + g1 + 4);
      const float4 fa = *(const float4*)(e2s + g0);
      const float4 fb = *(const float4*)(e2s + g0 + 4);
      const float4 fc = *(const float4*)(e2s + g1);
      const float4 fd = *(const float4*)(e2s + g1 + 4);
      const int   av[16] = {aa.x,aa.y,aa.z,aa.w, ab.x,ab.y,ab.z,ab.w,
                            ac.x,ac.y,ac.z,ac.w, ad.x,ad.y,ad.z,ad.w};
      const float ev[16] = {fa.x,fa.y,fa.z,fa.w, fb.x,fb.y,fb.z,fb.w,
                            fc.x,fc.y,fc.z,fc.w, fd.x,fd.y,fd.z,fd.w};
      unsigned bits = 0u;
#pragma unroll
      for (int i = 0; i < 16; ++i) {
        const bool on = av[i] > 0;
        const float t = __expf(lrelu02(e1v + ev[i]));
        sum += on ? t : 0.0f;
        bits |= (on ? 1u : 0u) << i;
      }
      wordbits |= bits << (kh * 16);
    }
    mlds[mbase + w] = wordbits;
    cnt += __popc(wordbits);
  }
  // combine the two lanes (L, L+16) that share a row
  sum += __shfl_xor(sum, 16);
  cnt += __shfl_xor(cnt, 16);                          // deg = row degree
  const float scale = (cnt > 0) ? ((float)cnt / sum) : 0.0f;  // all-masked row -> 0

  // ---- Pass B: P @ g with v_wmma_f32_16x16x32_bf16 -------------------------
  v8f acc[8] = {};
#pragma unroll 1
  for (int c = 0; c < 2; ++c) {                 // two 512-column chunks of g
    __syncthreads();
    // stage g chunk transposed into LDS as bf16: gt[d][m_local]
#pragma unroll 4
    for (int it = 0; it < 64; ++it) {
      const int f  = it * 256 + tid;            // 0..16383 float4s
      const int d4 = f & 31;                    // coalesced along d
      const int ml = f >> 5;                    // 0..511
      const float4 gv = *(const float4*)(g + (((size_t)b * NM + (c * 512 + ml)) * ND) + d4 * 4);
      const int dbase = d4 * 4;
      gt[(dbase + 0) * GT_STRIDE + ml] = (unsigned short)f2bf1(gv.x);
      gt[(dbase + 1) * GT_STRIDE + ml] = (unsigned short)f2bf1(gv.y);
      gt[(dbase + 2) * GT_STRIDE + ml] = (unsigned short)f2bf1(gv.z);
      gt[(dbase + 3) * GT_STRIDE + ml] = (unsigned short)f2bf1(gv.w);
    }
    __syncthreads();

#pragma unroll 1
    for (int kk = 0; kk < 16; ++kk) {
      const int k  = c * 16 + kk;
      const int g0 = k * 32 + moff;
      const int g1 = g0 + 16;
      const unsigned bits = (mlds[mbase + (k >> 1)] >> ((k & 1) * 16)) & 0xFFFFu;
      const float4 fa = *(const float4*)(e2s + g0);
      const float4 fb = *(const float4*)(e2s + g0 + 4);
      const float4 fc = *(const float4*)(e2s + g1);
      const float4 fd = *(const float4*)(e2s + g1 + 4);
      const float ev[16] = {fa.x,fa.y,fa.z,fa.w, fb.x,fb.y,fb.z,fb.w,
                            fc.x,fc.y,fc.z,fc.w, fd.x,fd.y,fd.z,fd.w};
      float p[16];
#pragma unroll
      for (int i = 0; i < 16; ++i) {
        const float t = __expf(lrelu02(e1v + ev[i])) * scale;
        p[i] = ((bits >> i) & 1u) ? t : 0.0f;
      }
      // A-frag: lane = row rit; elements = K (moff..moff+7, 16+moff..16+moff+7)
      v8u araw;
#pragma unroll
      for (int j = 0; j < 8; ++j) araw[j] = f2bf2(p[2 * j], p[2 * j + 1]);
      const v16bf afrag = __builtin_bit_cast(v16bf, araw);

      const int ml0 = kk * 32 + moff;           // chunk-local K base for this lane
#pragma unroll
      for (int t = 0; t < 8; ++t) {             // 8 N-tiles cover D=128
        const unsigned short* bp = &gt[(t * 16 + rit) * GT_STRIDE + ml0];
        const uint4 lo = *(const uint4*)(bp);        // K moff..moff+7
        const uint4 hi = *(const uint4*)(bp + 16);   // K 16+moff..16+moff+7
        v8u braw;
        braw[0] = lo.x; braw[1] = lo.y; braw[2] = lo.z; braw[3] = lo.w;
        braw[4] = hi.x; braw[5] = hi.y; braw[6] = hi.z; braw[7] = hi.w;
        const v16bf bfrag = __builtin_bit_cast(v16bf, braw);
        acc[t] = __builtin_amdgcn_wmma_f32_16x16x32_bf16(
            false, afrag, false, bfrag, (short)0, acc[t], false, false);
      }
    }
  }

  // ---- store C tiles: lane -> col, VGPR r -> row (upper lanes +8) ----------
#pragma unroll
  for (int t = 0; t < 8; ++t) {
    const int dcol = t * 16 + rit;
#pragma unroll
    for (int r = 0; r < 8; ++r) {
      const int n = n0 + r + half * 8;
      out[(((size_t)b * NN + n) * ND) + dcol] = acc[t][r];
    }
  }
}

// ---------------------------------------------------------------------------
extern "C" void kernel_launch(void* const* d_in, const int* in_sizes, int n_in,
                              void* d_out, int out_size, void* d_ws, size_t ws_size,
                              hipStream_t stream) {
  const float* h   = (const float*)d_in[0];   // (B,N,D)
  const float* g   = (const float*)d_in[1];   // (B,M,D)
  const int*   adj = (const int*)d_in[2];     // (B,N,M)
  const float* a1  = (const float*)d_in[3];   // (D,)
  const float* a2  = (const float*)d_in[4];   // (D,)
  float* out = (float*)d_out;                 // (B,N,D)

  float* e1 = (float*)d_ws;                   // B*N floats
  float* e2 = e1 + NB * NN;                   // B*M floats

  gat_edge_kernel<<<(NB * (NN + NM)) / 8, 256, 0, stream>>>(h, g, a1, a2, e1, e2);
  gat_attn_kernel<<<dim3(NN / 128, NB), 256, 0, stream>>>(g, adj, e1, e2, out);
}
